// AttentionFlow_34763465294206
// MI455X (gfx1250) — matrix operand order repaired and checked by reading, not compiled, not run on hardware
//
#include <hip/hip_runtime.h>
#include <hip/hip_bf16.h>

// Problem constants (reference: BS, T, N, DIM = 16, 2048, 2048, 128)
#define BSZ 16
#define TT  2048
#define NN  2048
#define DIM 128
#define NC  64     // n-columns processed per chunk

typedef __attribute__((ext_vector_type(16))) __bf16          v16bf;
typedef __attribute__((ext_vector_type(8)))  float           v8f;
typedef __attribute__((ext_vector_type(8)))  unsigned short  v8u16;

union FragAB {
    v16bf v;      // WMMA 16-bit A/B operand (8 VGPRs)
    v8u16 h[2];   // two 16-byte halves for LDS loads
};

__device__ __forceinline__ unsigned short f2bf(float f) {
    union { float f; unsigned u; } x; x.f = f;
    unsigned r = 0x7FFFu + ((x.u >> 16) & 1u);   // round-to-nearest-even
    return (unsigned short)((x.u + r) >> 16);
}

// packed f32x2 -> bf16x2 (one v_cvt_pk_bf16_f32 when available)
__device__ __forceinline__ unsigned cvtpk_bf16(float a, float b) {
#if __has_builtin(__builtin_amdgcn_cvt_pk_bf16_f32)
    auto v = __builtin_amdgcn_cvt_pk_bf16_f32(a, b);
    union { decltype(v) x; unsigned u; } cv; cv.x = v;
    return cv.u;
#else
    return (unsigned)f2bf(a) | ((unsigned)f2bf(b) << 16);
#endif
}

__device__ __forceinline__ unsigned long long pack2u32(unsigned lo, unsigned hi) {
    return (unsigned long long)lo | ((unsigned long long)hi << 32);
}

// ---------------------------------------------------------------------------
// Kernel 1: fused flash-style attention.
//   S = (aud*w3) @ sem^T + rowbias + colbias + b ; at = softmax_n(S)
//   h_con_w = at @ sem  -> out[...,128:256] ;  rowmax_n(S) -> Mws
// Block = 128 threads = 4 waves; each wave owns a 16-row t strip; the block
// marches over N in 64-column chunks (32 bf16 WMMAs per wave per chunk).
// ---------------------------------------------------------------------------
__global__ __launch_bounds__(128) void k1_flash(
    const float* __restrict__ aud, const float* __restrict__ sem,
    const float* __restrict__ W,   const float* __restrict__ bias,
    float* __restrict__ out, float* __restrict__ Mws)
{
    __shared__ float          s_w[3 * DIM];          // w1 | w2 | w3
    __shared__ float          s_qb[64];              // per-row bias: aud·w1 + b
    __shared__ unsigned short s_Q[64 * DIM];         // (aud*w3) bf16, [t][d]
    __shared__ unsigned short s_K[2][NC * DIM];      // sem bf16, [n][d]   (S-GEMM B)
    __shared__ unsigned short s_Kt[2][DIM * NC];     // sem bf16, [d][n]   (PV-GEMM B)
    __shared__ float          s_kb[2][NC];           // per-col bias: sem·w2
    __shared__ unsigned short s_P[4][16 * NC];       // per-wave P staging (C->A layout)

    const int tid = threadIdx.x;
    const int b   = blockIdx.x >> 5;      // 32 t-tiles per batch
    const int t0  = (blockIdx.x & 31) * 64;

    // ---- preload W (384 floats) ----
    s_w[tid] = W[tid]; s_w[tid + 128] = W[tid + 128]; s_w[tid + 256] = W[tid + 256];
    const float b0 = bias[0];
    __syncthreads();

    // ---- stage Q tile: rows t0..t0+63 ; 2 threads per row ----
    {
        const int row = tid >> 1;
        const int d0  = (tid & 1) * 64;
        const float* arow = aud + (size_t)(b * TT + t0 + row) * DIM + d0;
        float qdot = 0.f;
#pragma unroll
        for (int j = 0; j < 16; ++j) {
            float4 f = ((const float4*)arow)[j];
            const int d = d0 + j * 4;
            qdot += f.x * s_w[d] + f.y * s_w[d + 1] + f.z * s_w[d + 2] + f.w * s_w[d + 3];
            unsigned q01 = cvtpk_bf16(f.x * s_w[256 + d],     f.y * s_w[256 + d + 1]);
            unsigned q23 = cvtpk_bf16(f.z * s_w[256 + d + 2], f.w * s_w[256 + d + 3]);
            *(unsigned long long*)&s_Q[row * DIM + d] = pack2u32(q01, q23);
        }
        qdot += __shfl_xor(qdot, 1);
        if ((tid & 1) == 0) s_qb[row] = qdot + b0;
    }
    __syncthreads();

    const int lane  = tid & 31;
    const int wv    = tid >> 5;
    const int laneN = lane & 15;
    const int hi    = lane >> 4;

    // ---- Q A-fragments (bf16 16x32 layout, ISA 7.12.2): live for whole loop ----
    FragAB aQ[4];
#pragma unroll
    for (int kk = 0; kk < 4; ++kk) {
        const unsigned short* q = &s_Q[(wv * 16 + laneN) * DIM + kk * 32 + hi * 8];
        aQ[kk].h[0] = *(const v8u16*)q;          // K = hi*8 .. hi*8+7
        aQ[kk].h[1] = *(const v8u16*)(q + 16);   // K = 16+hi*8 ..
    }
    float qb_r[8];
#pragma unroll
    for (int r = 0; r < 8; ++r) qb_r[r] = s_qb[wv * 16 + r + 8 * hi];

    // ---- online-softmax state (C layout: row = r + 8*hi, col = laneN) ----
    // l is kept PER-LANE (sum over this lane's columns) and reduced only once
    // at the end: the corr rescale factor is row-uniform, so this is exact.
    float m_r[8], l_lane[8];
    v8f   accO[8];
    const v8f zero8 = {0.f, 0.f, 0.f, 0.f, 0.f, 0.f, 0.f, 0.f};
#pragma unroll
    for (int r = 0; r < 8; ++r) { m_r[r] = -3.0e38f; l_lane[r] = 0.f; }
#pragma unroll
    for (int dt = 0; dt < 8; ++dt) accO[dt] = zero8;

    // cooperative staging of a 64-row sem chunk: each thread owns 2 rows x 32 d
    // so the [d][n] transpose writes become packed 32-bit stores.
    auto stage = [&](int buf, int n0) {
        const int pr = tid >> 2;              // row pair 0..31 -> rows 2pr, 2pr+1
        const int d0 = (tid & 3) * 32;
        const float* r0 = sem + (size_t)(b * NN + n0 + 2 * pr) * DIM + d0;
        const float* r1 = r0 + DIM;
        if (n0 + 2 * NC < NN) __builtin_prefetch(r0 + (size_t)NC * DIM, 0, 0);
        float kd0 = 0.f, kd1 = 0.f;
#pragma unroll
        for (int j = 0; j < 8; ++j) {
            float4 f0 = ((const float4*)r0)[j];
            float4 f1 = ((const float4*)r1)[j];
            const int d = d0 + j * 4;
            kd0 += f0.x * s_w[128 + d]     + f0.y * s_w[128 + d + 1] +
                   f0.z * s_w[128 + d + 2] + f0.w * s_w[128 + d + 3];
            kd1 += f1.x * s_w[128 + d]     + f1.y * s_w[128 + d + 1] +
                   f1.z * s_w[128 + d + 2] + f1.w * s_w[128 + d + 3];
            // [n][d] rows, 8-byte packed stores
            *(unsigned long long*)&s_K[buf][(2 * pr) * DIM + d] =
                pack2u32(cvtpk_bf16(f0.x, f0.y), cvtpk_bf16(f0.z, f0.w));
            *(unsigned long long*)&s_K[buf][(2 * pr + 1) * DIM + d] =
                pack2u32(cvtpk_bf16(f1.x, f1.y), cvtpk_bf16(f1.z, f1.w));
            // [d][n] transpose, adjacent n pair packed into one 32-bit store
            *(unsigned*)&s_Kt[buf][(d + 0) * NC + 2 * pr] = cvtpk_bf16(f0.x, f1.x);
            *(unsigned*)&s_Kt[buf][(d + 1) * NC + 2 * pr] = cvtpk_bf16(f0.y, f1.y);
            *(unsigned*)&s_Kt[buf][(d + 2) * NC + 2 * pr] = cvtpk_bf16(f0.z, f1.z);
            *(unsigned*)&s_Kt[buf][(d + 3) * NC + 2 * pr] = cvtpk_bf16(f0.w, f1.w);
        }
        kd0 += __shfl_xor(kd0, 1); kd0 += __shfl_xor(kd0, 2);
        kd1 += __shfl_xor(kd1, 1); kd1 += __shfl_xor(kd1, 2);
        if ((tid & 3) == 0) { s_kb[buf][2 * pr] = kd0; s_kb[buf][2 * pr + 1] = kd1; }
    };

    stage(0, 0);
    int buf = 0;
    for (int c = 0; c < NN / NC; ++c) {
        __syncthreads();                               // staged chunk c visible
        if (c + 1 < NN / NC) stage(buf ^ 1, (c + 1) * NC);

        // ---- S tile (16 x 64): Q(16x128) @ K^T, 16 bf16 WMMAs ----
        v8f cc[4];
#pragma unroll
        for (int nt = 0; nt < 4; ++nt) cc[nt] = zero8;
#pragma unroll
        for (int kk = 0; kk < 4; ++kk) {
#pragma unroll
            for (int nt = 0; nt < 4; ++nt) {
                FragAB bK;
                const unsigned short* kp =
                    &s_K[buf][(nt * 16 + laneN) * DIM + kk * 32 + hi * 8];
                bK.h[0] = *(const v8u16*)kp;
                bK.h[1] = *(const v8u16*)(kp + 16);
                cc[nt] = __builtin_amdgcn_wmma_f32_16x16x32_bf16(
                    false, aQ[kk].v, false, bK.v, (short)0, cc[nt], false, false);
            }
        }

        float kb[4];
#pragma unroll
        for (int nt = 0; nt < 4; ++nt) kb[nt] = s_kb[buf][nt * 16 + laneN];

        // ---- online softmax over this 64-column chunk ----
        float corr_r[8];
#pragma unroll
        for (int r = 0; r < 8; ++r) {
            const float qb = qb_r[r];
            float s0 = cc[0][r] + qb + kb[0];
            float s1 = cc[1][r] + qb + kb[1];
            float s2 = cc[2][r] + qb + kb[2];
            float s3 = cc[3][r] + qb + kb[3];
            float cm = fmaxf(fmaxf(s0, s1), fmaxf(s2, s3));
            cm = fmaxf(cm, __shfl_xor(cm, 1));
            cm = fmaxf(cm, __shfl_xor(cm, 2));
            cm = fmaxf(cm, __shfl_xor(cm, 4));
            cm = fmaxf(cm, __shfl_xor(cm, 8));      // row max within 16-lane half
            const float nm   = fmaxf(m_r[r], cm);
            const float corr = __expf(m_r[r] - nm);
            const float p0 = __expf(s0 - nm);
            const float p1 = __expf(s1 - nm);
            const float p2 = __expf(s2 - nm);
            const float p3 = __expf(s3 - nm);
            l_lane[r] = l_lane[r] * corr + ((p0 + p1) + (p2 + p3));
            m_r[r]    = nm;
            corr_r[r] = corr;
            const int prow = r + 8 * hi;
            const unsigned pk01 = cvtpk_bf16(p0, p1);
            const unsigned pk23 = cvtpk_bf16(p2, p3);
            s_P[wv][prow * NC + laneN]      = (unsigned short)pk01;
            s_P[wv][prow * NC + 16 + laneN] = (unsigned short)(pk01 >> 16);
            s_P[wv][prow * NC + 32 + laneN] = (unsigned short)pk23;
            s_P[wv][prow * NC + 48 + laneN] = (unsigned short)(pk23 >> 16);
        }

#pragma unroll
        for (int dt = 0; dt < 8; ++dt) {
#pragma unroll
            for (int r = 0; r < 8; ++r) accO[dt][r] *= corr_r[r];
        }

        // ---- O += P(16x64) @ V(64x128): 16 bf16 WMMAs ----
        FragAB aP[2];
#pragma unroll
        for (int kc = 0; kc < 2; ++kc) {
            const unsigned short* pp = &s_P[wv][laneN * NC + kc * 32 + hi * 8];
            aP[kc].h[0] = *(const v8u16*)pp;
            aP[kc].h[1] = *(const v8u16*)(pp + 16);
        }
#pragma unroll
        for (int dt = 0; dt < 8; ++dt) {
#pragma unroll
            for (int kc = 0; kc < 2; ++kc) {
                FragAB bV;
                const unsigned short* vv =
                    &s_Kt[buf][(dt * 16 + laneN) * NC + kc * 32 + hi * 8];
                bV.h[0] = *(const v8u16*)vv;
                bV.h[1] = *(const v8u16*)(vv + 16);
                accO[dt] = __builtin_amdgcn_wmma_f32_16x16x32_bf16(
                    false, aP[kc].v, false, bV.v, (short)0, accO[dt], false, false);
            }
        }

        buf ^= 1;
    }

    // ---- epilogue: reduce per-lane l, write h_con_w + row max ----
#pragma unroll
    for (int r = 0; r < 8; ++r) {
        float l = l_lane[r];
        l += __shfl_xor(l, 1);
        l += __shfl_xor(l, 2);
        l += __shfl_xor(l, 4);
        l += __shfl_xor(l, 8);
        const float inv = 1.0f / l;
        const int   t   = t0 + wv * 16 + r + 8 * hi;
        float* orow = out + (size_t)(b * TT + t) * 512 + 128;
#pragma unroll
        for (int dt = 0; dt < 8; ++dt)
            orow[dt * 16 + laneN] = accO[dt][r] * inv;
        if (laneN == 0) Mws[b * TT + t] = m_r[r];
    }
}

// ---------------------------------------------------------------------------
// Kernel 2: bw = softmax_t(rowmax);  h_con_a2[b,d] = sum_t bw[t]*aud[b,t,d]
// ---------------------------------------------------------------------------
__global__ __launch_bounds__(256) void k2_bw(
    const float* __restrict__ aud, const float* __restrict__ Mws,
    float* __restrict__ h2ws)
{
    __shared__ float red[256];
    __shared__ float bwv[TT];
    const int tid = threadIdx.x;
    const int b   = blockIdx.x;
    const float* Mb = Mws + (size_t)b * TT;

    float mx = -3.0e38f;
    for (int t = tid; t < TT; t += 256) mx = fmaxf(mx, Mb[t]);
    red[tid] = mx; __syncthreads();
    for (int off = 128; off > 0; off >>= 1) {
        if (tid < off) red[tid] = fmaxf(red[tid], red[tid + off]);
        __syncthreads();
    }
    mx = red[0];
    __syncthreads();

    float sum = 0.f;
    for (int t = tid; t < TT; t += 256) {
        float e = __expf(Mb[t] - mx);
        bwv[t] = e;
        sum += e;
    }
    red[tid] = sum; __syncthreads();
    for (int off = 128; off > 0; off >>= 1) {
        if (tid < off) red[tid] += red[tid + off];
        __syncthreads();
    }
    const float inv = 1.0f / red[0];

    if (tid < DIM) {                                   // lane d; coalesced over d
        const float* ab = aud + (size_t)b * TT * DIM + tid;
        float acc = 0.f;
        for (int t = 0; t < TT; ++t) acc += bwv[t] * ab[(size_t)t * DIM];
        h2ws[b * DIM + tid] = acc * inv;
    }
}

// ---------------------------------------------------------------------------
// Kernel 3: assemble out = [aud | h_con_w | aud*h_con_w | aud*h_con_a2]
// (segment 1 was already written by kernel 1)
// ---------------------------------------------------------------------------
__global__ __launch_bounds__(256) void k3_assemble(
    const float* __restrict__ aud, const float* __restrict__ h2ws,
    float* __restrict__ out)
{
    const size_t i  = (size_t)blockIdx.x * 256 + threadIdx.x;  // over B*T*DIM
    const int    d  = (int)(i & (DIM - 1));
    const size_t bt = i >> 7;
    const int    b  = (int)(bt >> 11);
    const float  a  = aud[i];
    float* orow = out + bt * 512;
    const float h   = orow[128 + d];
    const float h2  = h2ws[b * DIM + d];
    orow[d]       = a;
    orow[256 + d] = a * h;
    orow[384 + d] = a * h2;
}

// ---------------------------------------------------------------------------
extern "C" void kernel_launch(void* const* d_in, const int* in_sizes, int n_in,
                              void* d_out, int out_size, void* d_ws, size_t ws_size,
                              hipStream_t stream)
{
    (void)in_sizes; (void)n_in; (void)out_size; (void)ws_size;
    const float* aud = (const float*)d_in[0];
    const float* sem = (const float*)d_in[1];
    const float* W   = (const float*)d_in[2];
    const float* bia = (const float*)d_in[3];
    float* out  = (float*)d_out;
    float* Mws  = (float*)d_ws;                 // BSZ*TT floats (row max of S)
    float* h2ws = Mws + (size_t)BSZ * TT;       // BSZ*DIM floats (h_con_a2)

    k1_flash   <<<BSZ * (TT / 64), 128, 0, stream>>>(aud, sem, W, bia, out, Mws);
    k2_bw      <<<BSZ,             256, 0, stream>>>(aud, Mws, h2ws);
    k3_assemble<<<(BSZ * TT * DIM) / 256, 256, 0, stream>>>(aud, h2ws, out);
}
